// EncoderLayer_37297495999146
// MI455X (gfx1250) — matrix operand order repaired and checked by reading, not compiled, not run on hardware
//
#include <hip/hip_runtime.h>

// ---------------------------------------------------------------------------
// Transformer encoder layer for MI455X (gfx1250, wave32, WMMA bf16->f32).
// B=2, S=2048, D_MODEL=1024, H=16, D_K=64, D_FF=2048.
// Weights pre-packed into WMMA-native k-blocked layout [K/32][N][32]; softmax
// row reductions in pure VALU DPP16; double-buffered async global->LDS
// staging (ASYNCcnt) overlapping DMA with WMMA compute.
// ---------------------------------------------------------------------------

typedef __attribute__((ext_vector_type(8)))  float          v8f;
typedef __attribute__((ext_vector_type(16))) __bf16         v16bf;
typedef __attribute__((ext_vector_type(8)))  unsigned short u16x8;
typedef __attribute__((ext_vector_type(4)))  int            v4i;

#define BATCH 2
#define SEQ 2048
#define DMODEL 1024
#define NHEAD 16
#define DHEAD 64
#define DFF 2048

// 1/sqrt(DHEAD) * log2(e): folded into Q so softmax can use exp2 directly.
#define QK_SCALE 0.1803368801111204f

#if defined(__has_builtin)
#if __has_builtin(__builtin_amdgcn_global_load_async_to_lds_b128)
#define HAVE_ASYNC_LDS 1
#endif
#if __has_builtin(__builtin_amdgcn_exp2f)
#define HAVE_EXP2 1
#endif
#endif

__device__ __forceinline__ unsigned short f32_to_bf16(float f) {
  return __builtin_bit_cast(unsigned short, (__bf16)f);
}

// Raw v_exp_f32: inputs are always <= 0 here; underflow-to-zero is desired.
__device__ __forceinline__ float fast_exp2(float x) {
#if defined(HAVE_EXP2)
  return __builtin_amdgcn_exp2f(x);
#else
  return exp2f(x);
#endif
}

// ---- copy 16B global -> LDS (async-to-LDS when available) -----------------
__device__ __forceinline__ void cp_b128(unsigned short* lds,
                                        const unsigned short* g) {
#if defined(HAVE_ASYNC_LDS)
  __builtin_amdgcn_global_load_async_to_lds_b128(
      (__attribute__((address_space(1))) v4i*)g,
      (__attribute__((address_space(3))) v4i*)lds, 0, 0);
#else
  *(u16x8*)lds = *(const u16x8*)g;
#endif
}
__device__ __forceinline__ void cp_wait() {
#if defined(HAVE_ASYNC_LDS)
#if __has_builtin(__builtin_amdgcn_s_wait_asynccnt)
  __builtin_amdgcn_s_wait_asynccnt(0);
#else
  asm volatile("s_wait_asynccnt 0x0" ::: "memory");
#endif
#endif
}

// ---- DPP16 16-lane butterfly reductions (keeps lane halves independent) ---
template <int CTRL>
__device__ __forceinline__ float dpp_f32(float v) {
  return __builtin_bit_cast(
      float, __builtin_amdgcn_update_dpp(0, __builtin_bit_cast(int, v), CTRL,
                                         0xF, 0xF, false));
}
__device__ __forceinline__ float red_max16(float v) {
  v = fmaxf(v, dpp_f32<0xB1>(v));   // quad_perm [1,0,3,2]  (xor 1)
  v = fmaxf(v, dpp_f32<0x4E>(v));   // quad_perm [2,3,0,1]  (xor 2)
  v = fmaxf(v, dpp_f32<0x141>(v));  // row_half_mirror      (xor 4 effect)
  v = fmaxf(v, dpp_f32<0x140>(v));  // row_mirror           (xor 8 effect)
  return v;
}
__device__ __forceinline__ float red_sum16(float v) {
  v += dpp_f32<0xB1>(v);
  v += dpp_f32<0x4E>(v);
  v += dpp_f32<0x141>(v);
  v += dpp_f32<0x140>(v);
  return v;
}

// A-fragment (16-bit 16x32): lane half holds k = b0a..b0a+7 and b0a+16..b0a+23
// (b0a = 0 / 8).  p points at element [row][b0a]; elements contiguous 8+8.
__device__ __forceinline__ v16bf ld_frag_a(const unsigned short* p) {
  u16x8 lo = *(const u16x8*)p;
  u16x8 hi = *(const u16x8*)(p + 16);
  return __builtin_bit_cast(
      v16bf, __builtin_shufflevector(lo, hi, 0, 1, 2, 3, 4, 5, 6, 7, 8, 9, 10,
                                     11, 12, 13, 14, 15));
}

// B-fragment (16-bit 32x16): lane half holds k = b0b..b0b+7 and b0b+16..b0b+23
// (b0b = 0 / 8), same order as A; elements contiguous 8+8 at p.
__device__ __forceinline__ v16bf ld_frag_b(const unsigned short* p) {
  u16x8 lo = *(const u16x8*)p;
  u16x8 hi = *(const u16x8*)(p + 16);
  return __builtin_bit_cast(
      v16bf, __builtin_shufflevector(lo, hi, 0, 1, 2, 3, 4, 5, 6, 7, 8, 9, 10,
                                     11, 12, 13, 14, 15));
}

__device__ __forceinline__ v8f wmma_bf16(v16bf a, v16bf b, v8f c) {
  return __builtin_amdgcn_wmma_f32_16x16x32_bf16(false, a, false, b, (short)0,
                                                 c, false, false);
}

// ---------------------------------------------------------------------------
// x: f32 -> bf16 row-major
// ---------------------------------------------------------------------------
__global__ void cvt_f32_bf16_kernel(const float* __restrict__ in,
                                    unsigned short* __restrict__ out, int n) {
  int i = blockIdx.x * blockDim.x + threadIdx.x;
  int stride = gridDim.x * blockDim.x;
  for (; i < n; i += stride) out[i] = f32_to_bf16(in[i]);
}

// ---------------------------------------------------------------------------
// Weight pack: f32 [K][N] row-major -> bf16 [K/32][N][32] (k-blocked, WMMA
// B-fragment native).  Runs once per weight; fused with conversion.
// ---------------------------------------------------------------------------
__global__ void pack_w_kernel(const float* __restrict__ src,
                              unsigned short* __restrict__ dst, int K, int N) {
  int i = blockIdx.x * blockDim.x + threadIdx.x;
  const int total = K * N;
  const int stride = gridDim.x * blockDim.x;
  for (; i < total; i += stride) {
    const int kk = i & 31;
    const int n = (i >> 5) % N;
    const int kt = i / (32 * N);
    dst[i] = f32_to_bf16(src[(size_t)(kt * 32 + kk) * N + n]);
  }
}

// ---------------------------------------------------------------------------
// GEMM: C[M,N] = A[M,K](bf16 row-major) * Bp[K/32][N][32](bf16 packed)
//       + bias[N]; optional ReLU / f32 residual / QK-scale;
//       f32 / bf16 / packed-V outputs.
// Block tile 128x128, 8 waves; wave tile 32x64 (2x4 WMMA tiles, 8 wmma/step).
// Double-buffered async staging: one barrier per k-step, DMA overlaps WMMA.
// ---------------------------------------------------------------------------
template <bool RELU, bool HASRES, bool OUTF, bool OUTB, bool VPACK, bool SCALEQ>
__global__ __launch_bounds__(256) void gemm_bf16_kernel(
    const unsigned short* __restrict__ A, const unsigned short* __restrict__ Bp,
    const float* __restrict__ bias, const float* __restrict__ residual,
    float* __restrict__ outF, unsigned short* __restrict__ outB, int M, int N,
    int K) {
  __shared__ __align__(16) unsigned short As[2][128][40];  // 32 k + 8 pad
  __shared__ __align__(16) unsigned short Bs[2][128][40];  // [n][32 k] + 8 pad

  const int t = threadIdx.x;
  const int lane = t & 31;
  const int w = t >> 5;
  const int bm = blockIdx.y * 128;
  const int bn = blockIdx.x * 128;
  const int wm = (w & 3) * 32;   // wave row strip (4 strips)
  const int wn = (w >> 2) * 64;  // wave col strip (2 strips)

  const int ar = t >> 2, ac = (t & 3) * 8;  // A loader: 2 rows/thread
  const int lr = lane & 15;
  const int b0a = (lane & 16) ? 8 : 0;
  const int b0b = (lane & 16) ? 8 : 0;

  auto stage = [&](int buf, int kt) {
    cp_b128(&As[buf][ar][ac], &A[(size_t)(bm + ar) * K + kt + ac]);
    cp_b128(&As[buf][ar + 64][ac], &A[(size_t)(bm + ar + 64) * K + kt + ac]);
    const unsigned short* bsrc = Bp + ((size_t)(kt >> 5) * N + bn) * 32;
    cp_b128(&Bs[buf][t >> 1][(t & 1) * 16], &bsrc[t * 16]);
    cp_b128(&Bs[buf][t >> 1][(t & 1) * 16 + 8], &bsrc[t * 16 + 8]);
  };

  v8f acc[2][4] = {};

  stage(0, 0);
  cp_wait();
  __syncthreads();

  for (int kt = 0; kt < K; kt += 32) {
    const int cur = (kt >> 5) & 1;
    if (kt + 32 < K) stage(cur ^ 1, kt + 32);  // async: overlaps with WMMAs

    v16bf af0 = ld_frag_a(&As[cur][wm + lr][b0a]);
    v16bf af1 = ld_frag_a(&As[cur][wm + 16 + lr][b0a]);
#pragma unroll
    for (int nt = 0; nt < 4; ++nt) {
      v16bf bf = ld_frag_b(&Bs[cur][wn + nt * 16 + lr][b0b]);
      acc[0][nt] = wmma_bf16(af0, bf, acc[0][nt]);
      acc[1][nt] = wmma_bf16(af1, bf, acc[1][nt]);
    }
    cp_wait();
    __syncthreads();
  }

  const int rr = (lane & 16) ? 8 : 0;
#pragma unroll
  for (int mi = 0; mi < 2; ++mi)
#pragma unroll
    for (int nt = 0; nt < 4; ++nt)
#pragma unroll
      for (int r = 0; r < 8; ++r) {
        const int row = bm + wm + mi * 16 + r + rr;
        const int col = bn + wn + nt * 16 + lr;
        const size_t o = (size_t)row * N + col;
        float v = acc[mi][nt][r] + bias[col];
        if (RELU) v = v > 0.f ? v : 0.f;
        if (HASRES) v += residual[o];
        if (SCALEQ) v *= QK_SCALE;
        if (OUTF) outF[o] = v;
        if (OUTB) outB[o] = f32_to_bf16(v);
        if (VPACK) {  // write V in per-head packed layout [b,h][S/32][64][32]
          const int b = row >> 11, key = row & (SEQ - 1);
          const int h = col >> 6, d = col & 63;
          const size_t po =
              ((((size_t)(b * NHEAD + h) * (SEQ / 32) + (key >> 5)) * 64 + d)
               << 5) +
              (key & 31);
          outB[po] = f32_to_bf16(v);
        }
      }
}

// ---------------------------------------------------------------------------
// Flash attention: block = (b, h, 128-query tile); 8 waves, one 16-query
// strip each.  64-key steps: S = Q*K^T (8 wmma), online softmax (raw v_exp,
// DPP16 row reductions), P via LDS relayout, O += P*V from packed global V
// (8 wmma).  K tiles double-buffered via async-to-LDS; Q pre-scaled by
// 1/sqrt(d)*log2(e) at projection time.
// ---------------------------------------------------------------------------
__global__ __launch_bounds__(256) void attn_kernel(
    const unsigned short* __restrict__ Qg, const unsigned short* __restrict__ Kg,
    const unsigned short* __restrict__ Vp, unsigned short* __restrict__ Og) {
  __shared__ __align__(16) unsigned short Qs[128][72];
  __shared__ __align__(16) unsigned short Ks[2][64][72];
  __shared__ __align__(16) unsigned short Ps[8][16][72];

  const int t = threadIdx.x, lane = t & 31, w = t >> 5;
  int idx = blockIdx.x;
  const int qb = idx % (SEQ / 128);
  idx /= (SEQ / 128);
  const int h = idx % NHEAD;
  const int b = idx / NHEAD;
  const int qbase = qb * 128;
  const size_t headoff = (size_t)h * DHEAD;
  const unsigned short* Vph =
      Vp + (size_t)(b * NHEAD + h) * (SEQ / 32) * DHEAD * 32;

  const int kvr = t >> 3, kvc = (t & 7) * 8;
  auto stageK = [&](int buf, int kb) {
#pragma unroll
    for (int p = 0; p < 2; ++p) {
      const int r = kvr + p * 32;
      cp_b128(&Ks[buf][r][kvc],
              &Kg[(size_t)(b * SEQ + kb + r) * DMODEL + headoff + kvc]);
    }
  };

  {  // stage Q tile (128 x 64) and first K tile
    const int r0 = t >> 3, c = (t & 7) * 8;
#pragma unroll
    for (int p = 0; p < 4; ++p) {
      const int r = r0 + p * 32;
      cp_b128(&Qs[r][c],
              &Qg[(size_t)(b * SEQ + qbase + r) * DMODEL + headoff + c]);
    }
  }
  stageK(0, 0);
  cp_wait();
  __syncthreads();

  const int lr = lane & 15;
  const int b0a = (lane & 16) ? 8 : 0;
  const int b0b = (lane & 16) ? 8 : 0;
  const int rr = (lane & 16) ? 8 : 0;

  const v16bf qf0 = ld_frag_a(&Qs[w * 16 + lr][b0a]);
  const v16bf qf1 = ld_frag_a(&Qs[w * 16 + lr][32 + b0a]);

  v8f oacc[4] = {};
  float mrow[8], lrow[8];
#pragma unroll
  for (int r = 0; r < 8; ++r) {
    mrow[r] = -3.0e38f;
    lrow[r] = 0.f;
  }

  for (int kb = 0; kb < SEQ; kb += 64) {
    const int cur = (kb >> 6) & 1;
    if (kb + 64 < SEQ) stageK(cur ^ 1, kb + 64);  // async: overlaps compute

    // scores: 16 queries x 64 keys (four 16-key n-tiles), k = d (2 halves)
    v8f sacc[4] = {};
#pragma unroll
    for (int nt = 0; nt < 4; ++nt) {
      v16bf kf0 = ld_frag_b(&Ks[cur][nt * 16 + lr][b0b]);
      v16bf kf1 = ld_frag_b(&Ks[cur][nt * 16 + lr][32 + b0b]);
      sacc[nt] = wmma_bf16(qf0, kf0, sacc[nt]);
      sacc[nt] = wmma_bf16(qf1, kf1, sacc[nt]);
    }

    // online softmax on C-layout (element r <-> row r / r+8 per lane half)
    float alpha[8];
#pragma unroll
    for (int r = 0; r < 8; ++r) {
      float mx = fmaxf(fmaxf(sacc[0][r], sacc[1][r]),
                       fmaxf(sacc[2][r], sacc[3][r]));
      mx = red_max16(mx);
      const float nm = fmaxf(mrow[r], mx);
      alpha[r] = fast_exp2(mrow[r] - nm);
      mrow[r] = nm;
      float rs = 0.f;
#pragma unroll
      for (int nt = 0; nt < 4; ++nt) {
        const float p = fast_exp2(sacc[nt][r] - nm);
        sacc[nt][r] = p;
        rs += p;
      }
      rs = red_sum16(rs);
      lrow[r] = lrow[r] * alpha[r] + rs;
    }

#pragma unroll
    for (int r = 0; r < 8; ++r) {
      oacc[0][r] *= alpha[r];
      oacc[1][r] *= alpha[r];
      oacc[2][r] *= alpha[r];
      oacc[3][r] *= alpha[r];
#pragma unroll
      for (int nt = 0; nt < 4; ++nt)
        Ps[w][r + rr][nt * 16 + lr] = f32_to_bf16(sacc[nt][r]);
    }

    // O += P (16x64) * V (64x64); P wave-local (LDS in-order within wave),
    // V fragments streamed from packed global (L2-resident).
    const v16bf pf0 = ld_frag_a(&Ps[w][lr][b0a]);
    const v16bf pf1 = ld_frag_a(&Ps[w][lr][32 + b0a]);
#pragma unroll
    for (int half = 0; half < 2; ++half) {
      const unsigned short* vbase =
          Vph + ((size_t)((kb >> 5) + half) * DHEAD) * 32;
#pragma unroll
      for (int nt = 0; nt < 4; ++nt) {
        v16bf vf = ld_frag_b(&vbase[(size_t)(nt * 16 + lr) * 32 + b0b]);
        oacc[nt] = wmma_bf16(half ? pf1 : pf0, vf, oacc[nt]);
      }
    }
    cp_wait();
    __syncthreads();  // next K tile landed; all waves done with cur
  }

#pragma unroll
  for (int nt = 0; nt < 4; ++nt)
#pragma unroll
    for (int r = 0; r < 8; ++r) {
      const int q = qbase + w * 16 + r + rr;
      const int d = nt * 16 + lr;
      const float v = oacc[nt][r] / lrow[r];
      Og[(size_t)(b * SEQ + q) * DMODEL + headoff + d] = f32_to_bf16(v);
    }
}

// ---------------------------------------------------------------------------
// LayerNorm over last dim (N=1024), one block (256 thr) per row.
// ---------------------------------------------------------------------------
template <bool OUTB>
__global__ __launch_bounds__(256) void layernorm_kernel(
    const float* __restrict__ x, const float* __restrict__ gamma,
    const float* __restrict__ beta, float* __restrict__ outF,
    unsigned short* __restrict__ outB) {
  __shared__ float red[256];
  const int row = blockIdx.x, t = threadIdx.x;
  const float* xr = x + (size_t)row * DMODEL;

  float local[4];
  float s = 0.f;
#pragma unroll
  for (int k = 0; k < 4; ++k) {
    local[k] = xr[t + k * 256];
    s += local[k];
  }
  red[t] = s;
  __syncthreads();
  for (int off = 128; off > 0; off >>= 1) {
    if (t < off) red[t] += red[t + off];
    __syncthreads();
  }
  const float mean = red[0] * (1.f / DMODEL);
  __syncthreads();

  float sq = 0.f;
#pragma unroll
  for (int k = 0; k < 4; ++k) {
    const float d = local[k] - mean;
    sq += d * d;
  }
  red[t] = sq;
  __syncthreads();
  for (int off = 128; off > 0; off >>= 1) {
    if (t < off) red[t] += red[t + off];
    __syncthreads();
  }
  const float rstd = rsqrtf(red[0] * (1.f / DMODEL) + 1e-5f);

#pragma unroll
  for (int k = 0; k < 4; ++k) {
    const int c = t + k * 256;
    const float y = (local[k] - mean) * rstd * gamma[c] + beta[c];
    const size_t o = (size_t)row * DMODEL + c;
    outF[o] = y;
    if (OUTB) outB[o] = f32_to_bf16(y);
  }
}

// ---------------------------------------------------------------------------
// Host orchestration
// ---------------------------------------------------------------------------
extern "C" void kernel_launch(void* const* d_in, const int* in_sizes, int n_in,
                              void* d_out, int out_size, void* d_ws,
                              size_t ws_size, hipStream_t stream) {
  (void)in_sizes; (void)n_in; (void)out_size; (void)ws_size;

  const float* x     = (const float*)d_in[0];
  const float* Wq    = (const float*)d_in[1];
  const float* bq    = (const float*)d_in[2];
  const float* Wk    = (const float*)d_in[3];
  const float* bk    = (const float*)d_in[4];
  const float* Wv    = (const float*)d_in[5];
  const float* bv    = (const float*)d_in[6];
  const float* Wo    = (const float*)d_in[7];
  const float* bo    = (const float*)d_in[8];
  const float* gamma = (const float*)d_in[9];
  const float* beta  = (const float*)d_in[10];
  const float* W1    = (const float*)d_in[11];
  const float* b1    = (const float*)d_in[12];
  const float* W2    = (const float*)d_in[13];
  const float* b2    = (const float*)d_in[14];

  const int M = BATCH * SEQ;  // 4096 rows
  const int D = DMODEL;

  char* ws = (char*)d_ws;
  size_t off = 0;
  auto alloc = [&](size_t bytes) -> void* {
    void* p = ws + off;
    off += (bytes + 255) & ~(size_t)255;
    return p;
  };
  unsigned short* xb   = (unsigned short*)alloc((size_t)M * D * 2);
  unsigned short* wqp  = (unsigned short*)alloc((size_t)D * D * 2);
  unsigned short* wkp  = (unsigned short*)alloc((size_t)D * D * 2);
  unsigned short* wvp  = (unsigned short*)alloc((size_t)D * D * 2);
  unsigned short* wop  = (unsigned short*)alloc((size_t)D * D * 2);
  unsigned short* w1p  = (unsigned short*)alloc((size_t)D * DFF * 2);
  unsigned short* w2p  = (unsigned short*)alloc((size_t)DFF * D * 2);
  unsigned short* qbuf = (unsigned short*)alloc((size_t)M * D * 2);
  unsigned short* kbuf = (unsigned short*)alloc((size_t)M * D * 2);
  unsigned short* vpk  = (unsigned short*)alloc((size_t)M * D * 2);
  unsigned short* attb = (unsigned short*)alloc((size_t)M * D * 2);
  float*          yf   = (float*)alloc((size_t)M * D * 4);
  float*          h1f  = (float*)alloc((size_t)M * D * 4);
  unsigned short* h1b  = (unsigned short*)alloc((size_t)M * D * 2);
  unsigned short* ff1b = (unsigned short*)alloc((size_t)M * DFF * 2);
  float*          ff2f = (float*)alloc((size_t)M * D * 4);

  const dim3 blk(256);

  // 1) convert x; pack all weights (fused f32->bf16 + k-blocked relayout)
  cvt_f32_bf16_kernel<<<2048, blk, 0, stream>>>(x, xb, M * D);
  pack_w_kernel<<<1024, blk, 0, stream>>>(Wq, wqp, D, D);
  pack_w_kernel<<<1024, blk, 0, stream>>>(Wk, wkp, D, D);
  pack_w_kernel<<<1024, blk, 0, stream>>>(Wv, wvp, D, D);
  pack_w_kernel<<<1024, blk, 0, stream>>>(Wo, wop, D, D);
  pack_w_kernel<<<2048, blk, 0, stream>>>(W1, w1p, D, DFF);
  pack_w_kernel<<<2048, blk, 0, stream>>>(W2, w2p, DFF, D);

  const dim3 gD(D / 128, M / 128);    // 8 x 32
  const dim3 gF(DFF / 128, M / 128);  // 16 x 32

  // 2) QKV projections (Q pre-scaled bf16, K row-major bf16, V packed)
  gemm_bf16_kernel<false, false, false, true, false, true>
      <<<gD, blk, 0, stream>>>(xb, wqp, bq, nullptr, nullptr, qbuf, M, D, D);
  gemm_bf16_kernel<false, false, false, true, false, false>
      <<<gD, blk, 0, stream>>>(xb, wkp, bk, nullptr, nullptr, kbuf, M, D, D);
  gemm_bf16_kernel<false, false, false, false, true, false>
      <<<gD, blk, 0, stream>>>(xb, wvp, bv, nullptr, nullptr, vpk, M, D, D);

  // 3) attention
  attn_kernel<<<BATCH * NHEAD * (SEQ / 128), blk, 0, stream>>>(qbuf, kbuf, vpk,
                                                               attb);

  // 4) output projection + residual (f32 out)
  gemm_bf16_kernel<false, true, true, false, false, false>
      <<<gD, blk, 0, stream>>>(attb, wop, bo, x, yf, nullptr, M, D, D);

  // 5) LayerNorm 1 -> h1 (f32 + bf16)
  layernorm_kernel<true><<<M, blk, 0, stream>>>(yf, gamma, beta, h1f, h1b);

  // 6) FFN: relu(h1 @ W1 + b1) -> bf16
  gemm_bf16_kernel<true, false, false, true, false, false>
      <<<gF, blk, 0, stream>>>(h1b, w1p, b1, nullptr, nullptr, ff1b, M, DFF, D);

  // 7) ff1 @ W2 + b2 + h1 residual -> f32
  gemm_bf16_kernel<false, true, true, false, false, false>
      <<<gD, blk, 0, stream>>>(ff1b, w2p, b2, h1f, ff2f, nullptr, M, D, DFF);

  // 8) LayerNorm 2 -> d_out (f32)
  layernorm_kernel<false><<<M, blk, 0, stream>>>(ff2f, gamma, beta,
                                                 (float*)d_out, nullptr);
}
